// Attention_50491635532232
// MI455X (gfx1250) — compile-verified, matrix operand order checked by reading, not run on hardware
//
#include <hip/hip_runtime.h>
#include <cstdint>
#include <cstddef>

// ---------------- problem constants ----------------
#define B_    2
#define S_    2048
#define DIM_  4096
#define NH_   32
#define NKV_  8
#define HD_   128
#define RANK_ 16
#define SCALE_ 2.0f            // ALPHA / RANK = 32/16
#define NTOK_ (B_ * S_)        // 4096 tokens
#define SOFTMAX_SCALE_ 0.08838834764831845f  // 1/sqrt(128)

typedef __bf16 bf16_t;
typedef __attribute__((ext_vector_type(16))) __bf16 v16bf;
typedef __attribute__((ext_vector_type(8)))  float  v8f;

// ---------------- async global->LDS (CDNA5), guarded ----------------
#if __has_builtin(__builtin_amdgcn_global_load_async_to_lds_b128)
#define ASYNC_LDS 1
typedef int v4i_vs __attribute__((vector_size(16)));
#define AS1_ __attribute__((address_space(1)))
#define AS3_ __attribute__((address_space(3)))
#endif

// copy 16 bytes (8 bf16) global -> LDS
__device__ __forceinline__ void cp16(bf16_t* l, const bf16_t* g) {
#ifdef ASYNC_LDS
  __builtin_amdgcn_global_load_async_to_lds_b128(
      (AS1_ v4i_vs*)g, (AS3_ v4i_vs*)l, 0, 0);
#else
  *(uint4*)l = *(const uint4*)g;
#endif
}

__device__ __forceinline__ void async_wait() {
#ifdef ASYNC_LDS
#if __has_builtin(__builtin_amdgcn_s_wait_asynccnt)
  __builtin_amdgcn_s_wait_asynccnt(0);
#else
  asm volatile("s_wait_asynccnt 0x0" ::: "memory");
#endif
#endif
}

// =====================================================================
// fp32 -> bf16 conversion (vectorized x4)
// =====================================================================
__global__ __launch_bounds__(256) void cvt_f32_bf16(const float* __restrict__ in,
                                                    bf16_t* __restrict__ out,
                                                    size_t n4) {
  size_t i = (size_t)blockIdx.x * blockDim.x + threadIdx.x;
  if (i >= n4) return;
  const float4 v = ((const float4*)in)[i];
  bf16_t* o = out + i * 4;
  o[0] = (bf16_t)v.x; o[1] = (bf16_t)v.y; o[2] = (bf16_t)v.z; o[3] = (bf16_t)v.w;
}

// =====================================================================
// WMMA GEMM:  C[M,N] (f32) = A[M,K] (bf16, row-major) * W[N,K]^T (bf16)
// Block tile 128x128x32; double-buffered LDS; 8 waves (2x4),
// wave tile 64(M) x 32(N) = 8 WMMAs / k-step.
// =====================================================================
#define BM 128
#define BN 128
#define BK 32

__global__ __launch_bounds__(256) void gemm_bf16_wmma(const bf16_t* __restrict__ A,
                                                      const bf16_t* __restrict__ W,
                                                      float* __restrict__ C,
                                                      int M, int N, int K) {
  __shared__ bf16_t As[2][BM][BK];   // 2 x 8 KB
  __shared__ bf16_t Bs[2][BN][BK];   // 2 x 8 KB

  const int tid  = threadIdx.x;
  const int wave = tid >> 5;
  const int lane = tid & 31;
  const int half = lane >> 4;
  const int l16  = lane & 15;
  const int kbase = half * 8;

  const int wm = wave >> 2;              // 0..1 -> M offset wm*64
  const int wn = wave & 3;               // 0..3 -> N offset wn*32
  const int row0 = wm * 64;
  const int col0 = wn * 32;
  const int blockM = blockIdx.y * BM;
  const int blockN = blockIdx.x * BN;

  // staging: 128 rows x 32 cols per tile; 256 thr x two 16B chunks
  const int srow = tid >> 1;             // 0..127
  const int scol = (tid & 1) * 16;       // 0,16
  const bf16_t* gA = A + (size_t)(blockM + srow) * K + scol;
  const bf16_t* gB = W + (size_t)(blockN + srow) * K + scol;

  v8f acc[4][2] = {};

  // prologue: stage tile 0 into buffer 0
  cp16(&As[0][srow][scol],     gA);
  cp16(&As[0][srow][scol + 8], gA + 8);
  cp16(&Bs[0][srow][scol],     gB);
  cp16(&Bs[0][srow][scol + 8], gB + 8);
  async_wait();
  __syncthreads();

  int cur = 0;
  for (int k0 = 0; k0 < K; k0 += BK) {
    // issue next tile's copies (direct-to-LDS, overlaps with WMMA below)
    if (k0 + BK < K) {
      const int nb = cur ^ 1;
      cp16(&As[nb][srow][scol],     gA + k0 + BK);
      cp16(&As[nb][srow][scol + 8], gA + k0 + BK + 8);
      cp16(&Bs[nb][srow][scol],     gB + k0 + BK);
      cp16(&Bs[nb][srow][scol + 8], gB + k0 + BK + 8);
    }

    v16bf af[4], bfr[2];
#pragma unroll
    for (int mi = 0; mi < 4; ++mi) {
      const int row = row0 + mi * 16 + l16;
#pragma unroll
      for (int e = 0; e < 16; ++e) {
        const int kk = (e >> 3) * 16 + kbase + (e & 7);
        af[mi][e] = As[cur][row][kk];
      }
    }
#pragma unroll
    for (int ni = 0; ni < 2; ++ni) {
      const int col = col0 + ni * 16 + l16;
#pragma unroll
      for (int e = 0; e < 16; ++e) {
        const int kk = (e >> 3) * 16 + kbase + (e & 7);
        bfr[ni][e] = Bs[cur][col][kk];
      }
    }
#pragma unroll
    for (int mi = 0; mi < 4; ++mi)
#pragma unroll
      for (int ni = 0; ni < 2; ++ni)
        acc[mi][ni] = __builtin_amdgcn_wmma_f32_16x16x32_bf16(
            false, af[mi], false, bfr[ni], (short)0, acc[mi][ni],
            false, false);

    async_wait();
    __syncthreads();
    cur ^= 1;
  }

#pragma unroll
  for (int mi = 0; mi < 4; ++mi)
#pragma unroll
    for (int ni = 0; ni < 2; ++ni)
#pragma unroll
      for (int r = 0; r < 8; ++r) {
        const int m = blockM + row0 + mi * 16 + half * 8 + r;
        const int n = blockN + col0 + ni * 16 + l16;
        C[(size_t)m * N + n] = acc[mi][ni][r];
      }
}

// =====================================================================
// LoRA down: t[token][r] = sum_k x[token][k] * a[r][k]   (rank 16, fp32)
// =====================================================================
__global__ __launch_bounds__(256) void lora_down(const float* __restrict__ x,
                                                 const float* __restrict__ a,
                                                 float* __restrict__ t, int K) {
  __shared__ float red[256];
  const int token = blockIdx.x;
  const float* xr = x + (size_t)token * K;
  for (int r = 0; r < RANK_; ++r) {
    float s = 0.f;
    for (int k = threadIdx.x; k < K; k += 256) s += xr[k] * a[(size_t)r * K + k];
    red[threadIdx.x] = s;
    __syncthreads();
    for (int off = 128; off > 0; off >>= 1) {
      if (threadIdx.x < off) red[threadIdx.x] += red[threadIdx.x + off];
      __syncthreads();
    }
    if (threadIdx.x == 0) t[(size_t)token * RANK_ + r] = red[0];
    __syncthreads();
  }
}

// LoRA up + add: C[token][n] += SCALE * sum_r t[token][r] * bm[n][r]
__global__ __launch_bounds__(256) void lora_up_add(float* __restrict__ C,
                                                   const float* __restrict__ t,
                                                   const float* __restrict__ bm,
                                                   int N) {
  const int n = blockIdx.x * 256 + threadIdx.x;
  const int token = blockIdx.y;
  if (n >= N) return;
  float s = 0.f;
#pragma unroll
  for (int r = 0; r < RANK_; ++r)
    s += t[(size_t)token * RANK_ + r] * bm[(size_t)n * RANK_ + r];
  C[(size_t)token * N + n] += SCALE_ * s;
}

// =====================================================================
// RoPE (interleaved pairs) + cast to bf16.  in/out: [B,S,H,HD]
// =====================================================================
__global__ __launch_bounds__(256) void rope_to_bf16(const float* __restrict__ in,
                                                    bf16_t* __restrict__ out,
                                                    const float* __restrict__ cosT,
                                                    const float* __restrict__ sinT,
                                                    int H) {
  const size_t total = (size_t)B_ * S_ * H * (HD_ / 2);
  size_t i = (size_t)blockIdx.x * blockDim.x + threadIdx.x;
  if (i >= total) return;
  const int d2 = (int)(i % (HD_ / 2));
  const size_t rest = i / (HD_ / 2);          // ((b*S + s)*H + h)
  const int s = (int)((rest / H) % S_);
  const size_t base = rest * HD_ + 2 * d2;
  const float c  = cosT[(size_t)s * (HD_ / 2) + d2];
  const float sn = sinT[(size_t)s * (HD_ / 2) + d2];
  const float tr = in[base], ti = in[base + 1];
  out[base]     = (bf16_t)(tr * c - ti * sn);
  out[base + 1] = (bf16_t)(tr * sn + ti * c);
}

// =====================================================================
// Flash attention (causal, GQA 4:1), bf16 WMMA, fp32 online softmax.
// Grid: (S/128, NH, B). Block: 256 = 8 waves; wave owns 16 q rows.
// K/V tiles staged cooperatively in LDS (V stored transposed so PV
// B-fragments are contiguous ds_load_b128).
// =====================================================================
__global__ __launch_bounds__(256) void flash_attn(const bf16_t* __restrict__ qb,
                                                  const bf16_t* __restrict__ kb,
                                                  const bf16_t* __restrict__ vb,
                                                  bf16_t* __restrict__ ob) {
  __shared__ bf16_t Ks[16][HD_];     // 4 KB: K rows j..j+15, row-major
  __shared__ bf16_t Vt[HD_][16];     // 4 KB: V transposed [d][kv]
  __shared__ bf16_t pS[8][16][16];   // 4 KB: per-wave P-tile bounce

  const int tid  = threadIdx.x;
  const int wave = tid >> 5;
  const int lane = tid & 31;
  const int half = lane >> 4;
  const int l16  = lane & 15;
  const int kbase = half * 8;

  const int bIdx = blockIdx.z;
  const int h    = blockIdx.y;
  const int kvh  = h / (NH_ / NKV_);
  const int blockQ0 = blockIdx.x * 128;
  const int qRow0   = blockQ0 + wave * 16;

  // staging indices: 16 rows x 128 cols = 256 thr x 8 elems (16B)
  const int srow = tid >> 4;              // kv row 0..15
  const int scol = (tid & 15) * 8;        // d col 0,8,...,120

  // ---- load Q fragments (16 rows x HD=128 -> 4 K-chunks of 32) ----
  v16bf qfrag[4];
  {
    const size_t qbase = (((size_t)bIdx * S_ + (qRow0 + l16)) * NH_ + h) * HD_;
#pragma unroll
    for (int c = 0; c < 4; ++c)
#pragma unroll
      for (int e = 0; e < 16; ++e) {
        const int d = c * 32 + (e >> 3) * 16 + kbase + (e & 7);
        qfrag[c][e] = qb[qbase + d];
      }
  }

  float rowMax[8], rowSum[8];
  v8f oacc[8] = {};
#pragma unroll
  for (int r = 0; r < 8; ++r) { rowMax[r] = -3.0e38f; rowSum[r] = 0.f; }

  const int kvEndBlock = blockQ0 + 128;   // uniform across block
  const int kvEndWave  = qRow0 + 16;      // this wave's causal limit

  for (int j = 0; j < kvEndBlock; j += 16) {
    // ---- cooperative staging of K (row-major) and V (transposed) ----
    {
      const size_t kvoff = (((size_t)bIdx * S_ + (j + srow)) * NKV_ + kvh) * HD_ + scol;
      cp16(&Ks[srow][scol], &kb[kvoff]);           // async-capable
      const uint4 vv = *(const uint4*)&vb[kvoff];  // manual transpose
      const bf16_t* pv = (const bf16_t*)&vv;
#pragma unroll
      for (int e = 0; e < 8; ++e) Vt[scol + e][srow] = pv[e];
    }
    async_wait();
    __syncthreads();

    if (j < kvEndWave) {
      // ---- S = Q K^T ----
      v8f sacc = {};
#pragma unroll
      for (int c = 0; c < 4; ++c) {
        v16bf kfrag;
#pragma unroll
        for (int e = 0; e < 16; ++e) {
          const int d = c * 32 + (e >> 3) * 16 + kbase + (e & 7);
          kfrag[e] = Ks[l16][d];
        }
        sacc = __builtin_amdgcn_wmma_f32_16x16x32_bf16(false, qfrag[c], false,
                                                       kfrag, (short)0, sacc,
                                                       false, false);
      }

      // ---- scale + causal mask + online softmax ----
      float pr[8];
#pragma unroll
      for (int r = 0; r < 8; ++r) {
        float v = sacc[r] * SOFTMAX_SCALE_;
        const int qg = qRow0 + half * 8 + r;
        const int kg = j + l16;
        if (kg > qg) v = -1.0e30f;

        float tm = v;                     // row max across 16 lanes
        tm = fmaxf(tm, __shfl_xor(tm, 1, 32));
        tm = fmaxf(tm, __shfl_xor(tm, 2, 32));
        tm = fmaxf(tm, __shfl_xor(tm, 4, 32));
        tm = fmaxf(tm, __shfl_xor(tm, 8, 32));

        const float nm   = fmaxf(rowMax[r], tm);
        const float corr = __expf(rowMax[r] - nm);
        rowMax[r] = nm;
        const float p = __expf(v - nm);
        float ps = p;
        ps += __shfl_xor(ps, 1, 32);
        ps += __shfl_xor(ps, 2, 32);
        ps += __shfl_xor(ps, 4, 32);
        ps += __shfl_xor(ps, 8, 32);
        rowSum[r] = rowSum[r] * corr + ps;
#pragma unroll
        for (int dt = 0; dt < 8; ++dt) oacc[dt][r] *= corr;
        pr[r] = p;
      }

      // ---- D-layout P -> A-layout bf16 fragment via per-wave LDS bounce ----
#pragma unroll
      for (int r = 0; r < 8; ++r)
        pS[wave][half * 8 + r][l16] = (bf16_t)pr[r];
      asm volatile("s_wait_dscnt 0" ::: "memory");  // order same-wave LDS RAW

      v16bf pfrag;
#pragma unroll
      for (int e = 0; e < 16; ++e) {
        const int kk = (e >> 3) * 16 + kbase + (e & 7);   // K index 0..31
        pfrag[e] = (kk < 16) ? pS[wave][l16][kk] : (bf16_t)0.0f;
      }

      // ---- O += P * V ----
#pragma unroll
      for (int dt = 0; dt < 8; ++dt) {
        v16bf vfrag;
        const int dcol = dt * 16 + l16;
#pragma unroll
        for (int e = 0; e < 16; ++e) {
          // kk = kbase + e%8 for e<8 (contiguous in Vt row), zero-pad e>=8
          vfrag[e] = (e < 8) ? Vt[dcol][kbase + e] : (bf16_t)0.0f;
        }
        oacc[dt] = __builtin_amdgcn_wmma_f32_16x16x32_bf16(false, pfrag, false,
                                                           vfrag, (short)0,
                                                           oacc[dt], false, false);
      }
    }
    __syncthreads();   // all reads done before next staging overwrites
  }

  // ---- normalize + store bf16 [B,S,NH,HD] ----
#pragma unroll
  for (int dt = 0; dt < 8; ++dt)
#pragma unroll
    for (int r = 0; r < 8; ++r) {
      const int qg = qRow0 + half * 8 + r;
      const int d  = dt * 16 + l16;
      ob[(((size_t)bIdx * S_ + qg) * NH_ + h) * HD_ + d] =
          (bf16_t)(oacc[dt][r] / rowSum[r]);
    }
}

// =====================================================================
// launch
// =====================================================================
extern "C" void kernel_launch(void* const* d_in, const int* in_sizes, int n_in,
                              void* d_out, int out_size, void* d_ws, size_t ws_size,
                              hipStream_t stream) {
  const float* x      = (const float*)d_in[0];
  const float* w_q    = (const float*)d_in[1];
  const float* w_k    = (const float*)d_in[2];
  const float* w_v    = (const float*)d_in[3];
  const float* w_o    = (const float*)d_in[4];
  const float* lq_a   = (const float*)d_in[5];
  const float* lq_b   = (const float*)d_in[6];
  const float* lk_a   = (const float*)d_in[7];
  const float* lk_b   = (const float*)d_in[8];
  const float* lv_a   = (const float*)d_in[9];
  const float* lv_b   = (const float*)d_in[10];
  const float* lo_a   = (const float*)d_in[11];
  const float* lo_b   = (const float*)d_in[12];
  const float* f_cos  = (const float*)d_in[15];
  const float* f_sin  = (const float*)d_in[16];
  float* out = (float*)d_out;

  // ---- workspace bump allocator ----
  char* ws = (char*)d_ws;
  size_t off = 0;
  auto alloc = [&](size_t bytes) -> void* {
    void* p = ws + off;
    off += (bytes + 255) & ~(size_t)255;
    return p;
  };
  const size_t NQ   = (size_t)NTOK_ * DIM_;          // 16.7M
  const size_t NKVE = (size_t)NTOK_ * NKV_ * HD_;    // 4.19M

  bf16_t* xb   = (bf16_t*)alloc(NQ * sizeof(bf16_t));
  bf16_t* wqb  = (bf16_t*)alloc(NQ * sizeof(bf16_t));
  bf16_t* wkb  = (bf16_t*)alloc((size_t)NKV_ * HD_ * DIM_ * sizeof(bf16_t));
  bf16_t* wvb  = (bf16_t*)alloc((size_t)NKV_ * HD_ * DIM_ * sizeof(bf16_t));
  bf16_t* wob  = (bf16_t*)alloc(NQ * sizeof(bf16_t));
  float*  qf   = (float*)alloc(NQ * sizeof(float));
  float*  kf   = (float*)alloc(NKVE * sizeof(float));
  float*  vf   = (float*)alloc(NKVE * sizeof(float));
  bf16_t* qbb  = (bf16_t*)alloc(NQ * sizeof(bf16_t));
  bf16_t* kbb  = (bf16_t*)alloc(NKVE * sizeof(bf16_t));
  bf16_t* vbb  = (bf16_t*)alloc(NKVE * sizeof(bf16_t));
  bf16_t* attb = (bf16_t*)alloc(NQ * sizeof(bf16_t));
  float*  tq   = (float*)alloc((size_t)NTOK_ * RANK_ * sizeof(float));
  float*  tk   = (float*)alloc((size_t)NTOK_ * RANK_ * sizeof(float));
  float*  tv   = (float*)alloc((size_t)NTOK_ * RANK_ * sizeof(float));
  float*  to   = (float*)alloc((size_t)NTOK_ * RANK_ * sizeof(float));
  (void)ws_size; (void)in_sizes; (void)n_in; (void)out_size;

  // ---- 1. bf16 conversions ----
  auto cvt = [&](const float* src, bf16_t* dst, size_t n) {
    size_t n4 = n / 4;
    cvt_f32_bf16<<<dim3((unsigned)((n4 + 255) / 256)), 256, 0, stream>>>(src, dst, n4);
  };
  cvt(x,   xb,  NQ);
  cvt(w_q, wqb, NQ);
  cvt(w_k, wkb, (size_t)NKV_ * HD_ * DIM_);
  cvt(w_v, wvb, (size_t)NKV_ * HD_ * DIM_);
  cvt(w_o, wob, NQ);

  // ---- 2. Q/K/V projections (WMMA) ----
  gemm_bf16_wmma<<<dim3(DIM_ / BN, NTOK_ / BM), 256, 0, stream>>>(
      xb, wqb, qf, NTOK_, DIM_, DIM_);
  gemm_bf16_wmma<<<dim3((NKV_ * HD_) / BN, NTOK_ / BM), 256, 0, stream>>>(
      xb, wkb, kf, NTOK_, NKV_ * HD_, DIM_);
  gemm_bf16_wmma<<<dim3((NKV_ * HD_) / BN, NTOK_ / BM), 256, 0, stream>>>(
      xb, wvb, vf, NTOK_, NKV_ * HD_, DIM_);

  // ---- 3. LoRA Q/K/V ----
  lora_down<<<NTOK_, 256, 0, stream>>>(x, lq_a, tq, DIM_);
  lora_down<<<NTOK_, 256, 0, stream>>>(x, lk_a, tk, DIM_);
  lora_down<<<NTOK_, 256, 0, stream>>>(x, lv_a, tv, DIM_);
  lora_up_add<<<dim3(DIM_ / 256, NTOK_), 256, 0, stream>>>(qf, tq, lq_b, DIM_);
  lora_up_add<<<dim3((NKV_ * HD_) / 256, NTOK_), 256, 0, stream>>>(kf, tk, lk_b, NKV_ * HD_);
  lora_up_add<<<dim3((NKV_ * HD_) / 256, NTOK_), 256, 0, stream>>>(vf, tv, lv_b, NKV_ * HD_);

  // ---- 4. RoPE + bf16 cast ----
  {
    size_t nq = (size_t)B_ * S_ * NH_ * (HD_ / 2);
    rope_to_bf16<<<dim3((unsigned)((nq + 255) / 256)), 256, 0, stream>>>(
        qf, qbb, f_cos, f_sin, NH_);
    size_t nk = (size_t)B_ * S_ * NKV_ * (HD_ / 2);
    rope_to_bf16<<<dim3((unsigned)((nk + 255) / 256)), 256, 0, stream>>>(
        kf, kbb, f_cos, f_sin, NKV_);
  }
  cvt(vf, vbb, NKVE);

  // ---- 5. causal flash attention (WMMA) ----
  flash_attn<<<dim3(S_ / 128, NH_, B_), 256, 0, stream>>>(qbb, kbb, vbb, attb);

  // ---- 6. O projection (WMMA) + LoRA O ----
  gemm_bf16_wmma<<<dim3(DIM_ / BN, NTOK_ / BM), 256, 0, stream>>>(
      attb, wob, out, NTOK_, DIM_, DIM_);
  lora_down<<<NTOK_, 256, 0, stream>>>(x, lo_a, to, DIM_);
  lora_up_add<<<dim3(DIM_ / 256, NTOK_), 256, 0, stream>>>(out, to, lo_b, DIM_);
}
